// VanillaRNN_3461743641060
// MI455X (gfx1250) — matrix operand order repaired and checked by reading, not compiled, not run on hardware
//
#include <hip/hip_runtime.h>
#include <hip/hip_bf16.h>

// ---------------------------------------------------------------------------
// 2-layer vanilla RNN, fully fused persistent kernel for MI455X (gfx1250).
//   batch=256, T=512, I=128, H=512, O=128
// 16 workgroups (one per 16-row batch tile) x 512 threads (16 wave32s).
// Each wave computes 32 of the 512 hidden columns with 2x v_wmma_f32_16x16x32_bf16
// accumulators; h-state lives in LDS (bf16, padded stride, conflict-free
// ds_load_b128 A-fragments); weights are pre-packed into B-fragment layout
// (contiguous 32B per lane) and stream from L2 every step.
//   * x-tile staged by the Tensor Data Mover (tensor_load_to_lds), double
//     buffered and synchronized with s_wait_tensorcnt (+ TDM LDS padding for
//     bank-conflict-free fp32 rows).
//   * tanh via v_tanh_f32 (or a 4-op exp-based fallback) instead of libm.
// ---------------------------------------------------------------------------

typedef __attribute__((ext_vector_type(16))) __bf16 v16bf;
typedef __attribute__((ext_vector_type(8)))  float  v8f;
typedef __attribute__((ext_vector_type(4)))  unsigned int u32x4;
typedef __attribute__((ext_vector_type(4)))  int i32x4;
typedef __attribute__((ext_vector_type(8)))  int i32x8;

#define RNN_B   256
#define RNN_T   512
#define RNN_I   128
#define RNN_H   512
#define RNN_O   128
#define BT      16          // batch rows per workgroup (WMMA M)
#define HPAD    520         // LDS bf16 row stride for H-wide tiles (bank pad)
#define XPAD    136         // LDS bf16 row stride for I-wide tile
#define XRAWPAD 132         // LDS fp32 row stride produced by TDM padding
#define OUT_HID0 (RNN_B * RNN_O)
#define OUT_HID1 (OUT_HID0 + RNN_B * RNN_H)

#if defined(__has_builtin)
#if __has_builtin(__builtin_amdgcn_tensor_load_to_lds)
#define HAVE_TDM 1
#endif
#endif
#ifndef HAVE_TDM
#define HAVE_TDM 0
#endif

__device__ __forceinline__ __bf16 f2bf(float f) {
  unsigned u = __float_as_uint(f);
  unsigned r = (u + 0x7FFFu + ((u >> 16) & 1u)) >> 16;
  unsigned short s = (unsigned short)r;
  return __builtin_bit_cast(__bf16, s);
}
__device__ __forceinline__ float bf2f(__bf16 b) {
  unsigned short s = __builtin_bit_cast(unsigned short, b);
  return __uint_as_float(((unsigned)s) << 16);
}

__device__ __forceinline__ float fast_tanh(float v) {
#if defined(__has_builtin)
#if __has_builtin(__builtin_amdgcn_tanhf)
  return __builtin_amdgcn_tanhf(v);          // v_tanh_f32
#else
  float e = __expf(2.0f * v);                // v_exp_f32
  return 1.0f - __fdividef(2.0f, e + 1.0f);  // v_rcp_f32 path
#endif
#else
  float e = __expf(2.0f * v);
  return 1.0f - __fdividef(2.0f, e + 1.0f);
#endif
}

#if HAVE_TDM
// Issue a TDM 2D tile load: 16 rows x 128 fp32, row stride T*I elements,
// into LDS at ldsOff with 4-dword padding after every 128-dword row
// (landed row stride = XRAWPAD dwords). Descriptor per CDNA5 ISA §8.3-8.6.
__device__ __forceinline__ void tdm_issue_x(unsigned ldsOff, unsigned long long gbyte) {
  u32x4 g0;
  g0[0] = 1u;                                          // count=1, user mode
  g0[1] = ldsOff;                                      // lds_addr (bytes)
  g0[2] = (unsigned)(gbyte & 0xffffffffull);           // global_addr[31:0]
  g0[3] = (unsigned)((gbyte >> 32) & 0x1ffffffull)     // global_addr[56:32]
          | (2u << 30);                                // type = 2 ("image")
  i32x8 g1;
  g1[0] = (2 << 16)                                    // data_size = 4 bytes
        | (1 << 20)                                    // pad_enable
        | (6 << 22)                                    // pad_interval = 128 dw
        | (3 << 25);                                   // pad_amount  = 4 dw
  g1[1] = (RNN_I & 0xffff) << 16;                      // tensor_dim0 = 128 (lo)
  g1[2] = (BT << 16);                                  // dim0 hi=0 | tensor_dim1 = 16 (lo)
  g1[3] = (RNN_I << 16);                               // dim1 hi=0 | tile_dim0 = 128
  g1[4] = BT;                                          // tile_dim1 = 16, tile_dim2 = 0
  g1[5] = RNN_T * RNN_I;                               // tensor_dim0_stride = 65536 (lo)
  g1[6] = 0;                                           // stride0 hi | stride1 lo
  g1[7] = 0;                                           // stride1 hi
  i32x4 g2; g2[0] = 1; g2[1] = 1; g2[2] = 0; g2[3] = 0; // dims 2/3 = 1, tile_dim3 = 0
  i32x4 g3; g3[0] = 0; g3[1] = 0; g3[2] = 0; g3[3] = 0;
#if __clang_major__ >= 23
  i32x8 g4 = {};
  __builtin_amdgcn_tensor_load_to_lds(g0, g1, g2, g3, g4, 0);
#else
  __builtin_amdgcn_tensor_load_to_lds(g0, g1, g2, g3, 0);
#endif
}
#endif

// Accumulate two 16x16 output tiles (columns nt0*16.., nt1*16..) over KT k-tiles.
// aRow: LDS pointer to this lane's A row start (+koff already applied).
// Bp:   packed weights; tile (nt*BKT+kt) is 32 lanes x 16 bf16 contiguous.
template <int KT, int BKT>
__device__ __forceinline__ void gemm_acc(v8f& c0, v8f& c1,
                                         const __bf16* aRow,
                                         const __bf16* Bp,
                                         int nt0, int nt1, int lane) {
#pragma unroll 4
  for (int kt = 0; kt < KT; ++kt) {
    union { v16bf v; u32x4 u[2]; } a, b0, b1;
    a.u[0] = *(const u32x4*)(aRow + kt * 32);
    a.u[1] = *(const u32x4*)(aRow + kt * 32 + 16);
    const __bf16* p0 = Bp + ((size_t)(nt0 * BKT + kt) * 32 + lane) * 16;
    const __bf16* p1 = Bp + ((size_t)(nt1 * BKT + kt) * 32 + lane) * 16;
    b0.u[0] = *(const u32x4*)p0;       b0.u[1] = *(const u32x4*)(p0 + 8);
    b1.u[0] = *(const u32x4*)p1;       b1.u[1] = *(const u32x4*)(p1 + 8);
    c0 = __builtin_amdgcn_wmma_f32_16x16x32_bf16(false, a.v, false, b0.v,
                                                 (short)0, c0, false, false);
    c1 = __builtin_amdgcn_wmma_f32_16x16x32_bf16(false, a.v, false, b1.v,
                                                 (short)0, c1, false, false);
  }
}

// Pack fp32 weight W[N][K] (row-major) into bf16 B-fragment layout:
// tile (nt*KT+kt): 32 lanes x 16 bf16; lane l holds N=nt*16+(l&15), K=kt*32+8*(l>>4)+(j&7)+16*(j>>3), j=0..15.
__global__ void rnn_pack_w(const float* __restrict__ W, __bf16* __restrict__ dst,
                           int N, int K) {
  int KT = K >> 5;
  int total = (N >> 4) * KT * 32;
  int tid = blockIdx.x * blockDim.x + threadIdx.x;
  if (tid >= total) return;
  int lane = tid & 31;
  int tile = tid >> 5;
  int kt = tile % KT;
  int nt = tile / KT;
  int k = kt * 32 + ((lane >> 4) << 3);
  __bf16* o = dst + (size_t)tile * 512 + (size_t)lane * 16;
#pragma unroll
  for (int j = 0; j < 16; ++j)
    o[j] = f2bf(W[(size_t)(nt * 16 + (lane & 15)) * K + k + (j & 7) + ((j >> 3) << 4)]);
}

__launch_bounds__(512, 1)
__global__ void rnn_fused(const float* __restrict__ x,
                          const float* __restrict__ hidden,
                          const float* __restrict__ b_ih0, const float* __restrict__ b_hh0,
                          const float* __restrict__ b_ih1, const float* __restrict__ b_hh1,
                          const float* __restrict__ fc_b,
                          const __bf16* __restrict__ Wih0p, const __bf16* __restrict__ Whh0p,
                          const __bf16* __restrict__ Wih1p, const __bf16* __restrict__ Whh1p,
                          const __bf16* __restrict__ fcWp,
                          float* __restrict__ out) {
  __shared__ __align__(16) __bf16 hA0[BT * HPAD];   // layer-0 h (also out0[t])
  __shared__ __align__(16) __bf16 hA1[BT * HPAD];   // layer-1 h
  __shared__ __align__(16) __bf16 xA[BT * XPAD];    // bf16 x tile at step t
#if HAVE_TDM
  __shared__ __align__(16) float xraw[2][BT * XRAWPAD];  // TDM-landed fp32 tiles
#endif
  __shared__ float bias0[RNN_H];
  __shared__ float bias1[RNN_H];

  const int tid    = threadIdx.x;
  const int wave   = tid >> 5;
  const int lane   = tid & 31;
  const int wgBase = blockIdx.x * BT;       // first batch row of this tile
  const int row    = lane & 15;             // A-fragment row (M)
  const int koff   = (lane < 16) ? 0 : 8;   // A-fragment K offset per ISA layout
  const int mbase  = (lane < 16) ? 0 : 8;   // C/D fragment M base per ISA layout
  const int nt0    = wave * 2;
  const int nt1    = wave * 2 + 1;
  const int n0     = nt0 * 16 + row;
  const int n1     = nt1 * 16 + row;

  // --- init: combined biases + initial hidden state (bf16 in LDS) ---
  for (int i = tid; i < RNN_H; i += 512) {
    bias0[i] = b_ih0[i] + b_hh0[i];
    bias1[i] = b_ih1[i] + b_hh1[i];
  }
  for (int i = tid; i < BT * RNN_H; i += 512) {
    int r = i >> 9, c = i & (RNN_H - 1);
    hA0[r * HPAD + c] = f2bf(hidden[(size_t)(wgBase + r) * RNN_H + c]);
    hA1[r * HPAD + c] = f2bf(hidden[(size_t)RNN_B * RNN_H + (size_t)(wgBase + r) * RNN_H + c]);
  }

  // x-tile addressing for this thread: 2048 elems / 512 threads = 4 each
  const int xi = tid * 4;
  const int xr = xi >> 7;            // row 0..15
  const int xc = xi & (RNN_I - 1);   // col 0..124 step 4

#if HAVE_TDM
  const unsigned long long xbase =
      (unsigned long long)(size_t)(const void*)x + (unsigned long long)wgBase * RNN_T * RNN_I * 4ull;
  if (wave == 0)   // prologue: async-load x tile for t=0
    tdm_issue_x((unsigned)(size_t)(void*)&xraw[0][0], xbase);
#endif
  __syncthreads();

  for (int t = 0; t < RNN_T; ++t) {
    // ---- stage x[:, t, :] tile -> bf16 LDS ----
#if HAVE_TDM
    if (wave == 0) {
      if (t + 1 < RNN_T) {  // async-load next step's tile into the other buffer
        tdm_issue_x((unsigned)(size_t)(void*)&xraw[(t + 1) & 1][0],
                    xbase + (unsigned long long)(t + 1) * RNN_I * 4ull);
        __builtin_amdgcn_s_wait_tensorcnt(1);  // step-t tile landed, t+1 in flight
      } else {
        __builtin_amdgcn_s_wait_tensorcnt(0);  // last tile landed
      }
    }
    __syncthreads();       // TDM data for step t visible to all waves
    {
      const float* s = &xraw[t & 1][xr * XRAWPAD + xc];
      __bf16* d = &xA[xr * XPAD + xc];
      d[0] = f2bf(s[0]); d[1] = f2bf(s[1]); d[2] = f2bf(s[2]); d[3] = f2bf(s[3]);
    }
#else
    {
      const float* xp = x + ((size_t)(wgBase + xr) * RNN_T + t) * RNN_I + xc;
      float4 v = *(const float4*)xp;
      __bf16* d = &xA[xr * XPAD + xc];
      d[0] = f2bf(v.x); d[1] = f2bf(v.y); d[2] = f2bf(v.z); d[3] = f2bf(v.w);
      if (t + 1 < RNN_T)
        __builtin_prefetch(xp + RNN_I, 0, 3);   // global_prefetch_b8
    }
#endif
    __syncthreads();

    // ---- layer 0: h0 = tanh(x@Wih0^T + h0@Whh0^T + bias0) ----
    v8f c0 = {}, c1 = {};
    gemm_acc<RNN_I / 32, RNN_I / 32>(c0, c1, &xA[row * XPAD + koff], Wih0p, nt0, nt1, lane);
    gemm_acc<RNN_H / 32, RNN_H / 32>(c0, c1, &hA0[row * HPAD + koff], Whh0p, nt0, nt1, lane);
    __syncthreads();   // all waves done reading hA0
    {
      float bb0 = bias0[n0], bb1 = bias0[n1];
#pragma unroll
      for (int r = 0; r < 8; ++r) {
        hA0[(mbase + r) * HPAD + n0] = f2bf(fast_tanh(c0[r] + bb0));
        hA0[(mbase + r) * HPAD + n1] = f2bf(fast_tanh(c1[r] + bb1));
      }
    }
    __syncthreads();   // hA0 now holds out0[t]

    // ---- layer 1: h1 = tanh(out0@Wih1^T + h1@Whh1^T + bias1) ----
    v8f d0 = {}, d1 = {};
    gemm_acc<RNN_H / 32, RNN_H / 32>(d0, d1, &hA0[row * HPAD + koff], Wih1p, nt0, nt1, lane);
    gemm_acc<RNN_H / 32, RNN_H / 32>(d0, d1, &hA1[row * HPAD + koff], Whh1p, nt0, nt1, lane);
    __syncthreads();   // all waves done reading hA1
    {
      float bb0 = bias1[n0], bb1 = bias1[n1];
#pragma unroll
      for (int r = 0; r < 8; ++r) {
        hA1[(mbase + r) * HPAD + n0] = f2bf(fast_tanh(d0[r] + bb0));
        hA1[(mbase + r) * HPAD + n1] = f2bf(fast_tanh(d1[r] + bb1));
      }
    }
    __syncthreads();
  }

  // ---- epilogue 1: new_hidden = [h0_last; h1_last] (fp32) ----
  for (int i = tid; i < BT * RNN_H; i += 512) {
    int r = i >> 9, c = i & (RNN_H - 1);
    out[OUT_HID0 + (size_t)(wgBase + r) * RNN_H + c] = bf2f(hA0[r * HPAD + c]);
    out[OUT_HID1 + (size_t)(wgBase + r) * RNN_H + c] = bf2f(hA1[r * HPAD + c]);
  }

  // ---- epilogue 2: output = h1_last @ fc_W^T + fc_b  (waves 0..7) ----
  if (wave < 8) {
    v8f c = {};
    const __bf16* aRow = &hA1[row * HPAD + koff];
#pragma unroll 4
    for (int kt = 0; kt < RNN_H / 32; ++kt) {
      union { v16bf v; u32x4 u[2]; } a, b;
      a.u[0] = *(const u32x4*)(aRow + kt * 32);
      a.u[1] = *(const u32x4*)(aRow + kt * 32 + 16);
      const __bf16* p = fcWp + ((size_t)(wave * (RNN_H / 32) + kt) * 32 + lane) * 16;
      b.u[0] = *(const u32x4*)p;
      b.u[1] = *(const u32x4*)(p + 8);
      c = __builtin_amdgcn_wmma_f32_16x16x32_bf16(false, a.v, false, b.v,
                                                  (short)0, c, false, false);
    }
    int n = (wave << 4) + row;
    float bb = fc_b[n];
#pragma unroll
    for (int r = 0; r < 8; ++r)
      out[(size_t)(wgBase + mbase + r) * RNN_O + n] = c[r] + bb;
  }
}

extern "C" void kernel_launch(void* const* d_in, const int* in_sizes, int n_in,
                              void* d_out, int out_size, void* d_ws, size_t ws_size,
                              hipStream_t stream) {
  const float* x      = (const float*)d_in[0];
  const float* hidden = (const float*)d_in[1];
  const float* W_ih0  = (const float*)d_in[2];
  const float* b_ih0  = (const float*)d_in[3];
  const float* W_hh0  = (const float*)d_in[4];
  const float* b_hh0  = (const float*)d_in[5];
  const float* W_ih1  = (const float*)d_in[6];
  const float* b_ih1  = (const float*)d_in[7];
  const float* W_hh1  = (const float*)d_in[8];
  const float* b_hh1  = (const float*)d_in[9];
  const float* fc_W   = (const float*)d_in[10];
  const float* fc_b   = (const float*)d_in[11];
  float* out = (float*)d_out;

  // packed bf16 weight layouts in workspace
  char* ws = (char*)d_ws;
  __bf16* Wih0p = (__bf16*)(ws + 0);         // 512x128  -> 131072 B
  __bf16* Whh0p = (__bf16*)(ws + 131072);    // 512x512  -> 524288 B
  __bf16* Wih1p = (__bf16*)(ws + 655360);    // 512x512
  __bf16* Whh1p = (__bf16*)(ws + 1179648);   // 512x512
  __bf16* fcWp  = (__bf16*)(ws + 1703936);   // 128x512  -> 131072 B

  // pack kernels: N*K/16 threads each
  rnn_pack_w<<<(512 * 128 / 16 + 255) / 256, 256, 0, stream>>>(W_ih0, Wih0p, 512, 128);
  rnn_pack_w<<<(512 * 512 / 16 + 255) / 256, 256, 0, stream>>>(W_hh0, Whh0p, 512, 512);
  rnn_pack_w<<<(512 * 512 / 16 + 255) / 256, 256, 0, stream>>>(W_ih1, Wih1p, 512, 512);
  rnn_pack_w<<<(512 * 512 / 16 + 255) / 256, 256, 0, stream>>>(W_hh1, Whh1p, 512, 512);
  rnn_pack_w<<<(128 * 512 / 16 + 255) / 256, 256, 0, stream>>>(fc_W, fcWp, 128, 512);

  rnn_fused<<<RNN_B / BT, 512, 0, stream>>>(x, hidden,
                                            b_ih0, b_hh0, b_ih1, b_hh1, fc_b,
                                            Wih0p, Whh0p, Wih1p, Whh1p, fcWp,
                                            out);
}
